// FutureInteractionModule_47880295416409
// MI455X (gfx1250) — compile-verified, matrix operand order checked by reading, not run on hardware
//
#include <hip/hip_runtime.h>

typedef __attribute__((ext_vector_type(16))) _Float16 v16h;
typedef __attribute__((ext_vector_type(8)))  _Float16 v8h;
typedef __attribute__((ext_vector_type(8)))  float    v8f;

#define NN 4096
#define TT 50
#define FF 16
#define HH 256
#define NHEAD 8
#define HC_ 8
#define EE 65536

#define WMMA(a, b, c) __builtin_amdgcn_wmma_f32_16x16x32_f16(false, (a), false, (b), (short)0, (c), false, false)

// ---------- WMMA fragment loaders (CDNA5 16x16x32 f16 layouts, ISA 7.12.2) ----------
__device__ __forceinline__ v16h frag_a(const _Float16* A, int lda, int row0, int k0, int lane) {
  int r = row0 + (lane & 15);
  int g = (lane >> 4) & 1;
  const _Float16* base = A + (size_t)r * lda + k0 + 8 * g;
  v16h f;
#pragma unroll
  for (int v = 0; v < 8; ++v) {
    int k = ((v & 3) << 1) + ((v >> 2) << 4);
    f[2 * v]     = base[k];
    f[2 * v + 1] = base[k + 1];
  }
  return f;
}

// B transposed in memory: element (k, n) = Bt[n*ldt + k] -> contiguous per lane
__device__ __forceinline__ v16h frag_bt(const _Float16* Bt, int ldt, int n0, int lane) {
  int n = n0 + (lane & 15);
  int g = (lane >> 4) & 1;
  const _Float16* base = Bt + (size_t)n * ldt + 8 * g;
  v16h f;
#pragma unroll
  for (int v = 0; v < 8; ++v) {
    int k = ((v & 3) << 1) + ((v >> 2) << 4);
    f[2 * v]     = base[k];
    f[2 * v + 1] = base[k + 1];
  }
  return f;
}

// ---------- swizzled (fragment-major) B panels ----------
__device__ __forceinline__ size_t bswz_idx(int k, int c, int ntn) {
  int kt = k >> 5, kr = k & 31;
  int nt = c >> 4, nr = c & 15;
  int half = kr & 1;
  int v = ((kr >> 1) & 3) | (((kr >> 4) & 1) << 2);
  int g = (kr >> 3) & 1;
  int lane = (g << 4) | nr;
  return ((((size_t)kt * ntn + nt) * 32) + lane) * 16 + (v << 1) + half;
}

__device__ __forceinline__ v16h frag_b_swz(const _Float16* Bs, int ntn, int k0, int col0, int lane) {
  const _Float16* p = Bs + ((((size_t)(k0 >> 5) * ntn + (col0 >> 4)) * 32) + lane) * 16;
  return *(const v16h*)p;  // 32B contiguous -> 2x b128
}

// ---------- small utility kernels ----------
__global__ void fill_f32(float* p, float v, size_t n) {
  for (size_t i = (size_t)blockIdx.x * blockDim.x + threadIdx.x; i < n; i += (size_t)gridDim.x * blockDim.x)
    p[i] = v;
}
__global__ void fill_f16(_Float16* p, size_t n) {
  for (size_t i = (size_t)blockIdx.x * blockDim.x + threadIdx.x; i < n; i += (size_t)gridDim.x * blockDim.x)
    p[i] = (_Float16)0.f;
}

__global__ void wtrans_swz(const float* __restrict__ W, _Float16* __restrict__ B,
                           int outd, int ind, int ntn, int rowoff) {
  size_t total = (size_t)outd * ind;
  for (size_t i = (size_t)blockIdx.x * blockDim.x + threadIdx.x; i < total; i += (size_t)gridDim.x * blockDim.x) {
    int k = (int)(i / outd), c = (int)(i % outd);
    B[bswz_idx(rowoff + k, c, ntn)] = (_Float16)W[(size_t)c * ind + k];
  }
}

__global__ void xpad_k(const float* __restrict__ x, _Float16* __restrict__ xp) {
  size_t total = (size_t)TT * NN * 32;
  for (size_t i = (size_t)blockIdx.x * blockDim.x + threadIdx.x; i < total; i += (size_t)gridDim.x * blockDim.x) {
    int k = (int)(i & 31);
    size_t nt = i >> 5;
    int n = (int)(nt % NN), t = (int)(nt / NN);
    xp[i] = (k < FF) ? (_Float16)x[((size_t)n * TT + t) * FF + k] : (_Float16)0.f;
  }
}

// vt[(head*32+c)*NN + n] = qkv[n*768 + 512 + head*32 + c]
__global__ void vtrans(const _Float16* __restrict__ qkv, _Float16* __restrict__ vt) {
  size_t total = (size_t)256 * NN;
  for (size_t i = (size_t)blockIdx.x * blockDim.x + threadIdx.x; i < total; i += (size_t)gridDim.x * blockDim.x) {
    int hc = (int)(i / NN), n = (int)(i % NN);
    vt[i] = qkv[(size_t)n * 768 + 512 + hc];
  }
}

// ---------- generic WMMA GEMM: C = A(MxK,f16) @ Bswz(KxN,f16) + bias ----------
template <int K, bool WF32, bool WF16>
__global__ __launch_bounds__(256) void gemm_wmma(const _Float16* __restrict__ A,
                                                 const _Float16* __restrict__ Bs,
                                                 const float* __restrict__ bias,
                                                 float* __restrict__ Cf, _Float16* __restrict__ Ch,
                                                 int M, int Nc) {
  int wave = threadIdx.x >> 5, lane = threadIdx.x & 31;
  int ntn = Nc >> 4;
  int tile = blockIdx.x * 8 + wave;
  if (tile >= (M >> 4) * ntn) return;
  int tm = tile / ntn, tn = tile - tm * ntn;
  v8f acc = {};
#pragma unroll
  for (int k = 0; k < K; k += 32) {
    if (k + 32 < K) __builtin_prefetch(A + (size_t)(tm << 4) * K + k + 32, 0, 0);
    v16h a = frag_a(A, K, tm << 4, k, lane);
    v16h b = frag_b_swz(Bs, ntn, k, tn << 4, lane);
    acc = WMMA(a, b, acc);
  }
  int c = (tn << 4) + (lane & 15);
  size_t off = (size_t)((tm << 4) + (((lane >> 4) & 1) << 3)) * Nc + c;
  float bv = bias[c];
#pragma unroll
  for (int v = 0; v < 8; ++v) {
    float o = acc[v] + bv;
    if (WF32) Cf[off] = o;
    if (WF16) Ch[off] = (_Float16)o;
    off += Nc;
  }
}

// ---------- fused LSTM step, no LDS ----------
// WG covers 32 rows; each wave owns a 32-wide hidden block across all 4 gates and computes
// TWO 16-row tiles, reusing every B fragment from registers twice (halves L2 B-panel traffic).
template <int K0>
__global__ __launch_bounds__(256) void lstm_cell(const _Float16* __restrict__ A0, int lda0,
                                                 const _Float16* __restrict__ Bs,  // swizzled (K0+256) x 1024
                                                 const float* __restrict__ b_ih, const float* __restrict__ b_hh,
                                                 float* __restrict__ cst, _Float16* __restrict__ h) {
  int wave = threadIdx.x >> 5, lane = threadIdx.x & 31;
  int row0 = blockIdx.x << 5;  // 32 rows per WG
  int j0 = wave << 5;          // hidden columns [j0, j0+32)
  v8f acc[16] = {};            // [rowtile*8 + gate*2 + colhalf]
  for (int k = 0; k < K0; k += 32) {
    v16h a0 = frag_a(A0, lda0, row0, k, lane);
    v16h a1 = frag_a(A0, lda0, row0 + 16, k, lane);
#pragma unroll
    for (int gq = 0; gq < 4; ++gq)
#pragma unroll
      for (int b = 0; b < 2; ++b) {
        v16h bf = frag_b_swz(Bs, 64, k, gq * 256 + j0 + b * 16, lane);
        acc[gq * 2 + b]     = WMMA(a0, bf, acc[gq * 2 + b]);
        acc[8 + gq * 2 + b] = WMMA(a1, bf, acc[8 + gq * 2 + b]);
      }
  }
  for (int k = 0; k < 256; k += 32) {
    v16h a0 = frag_a(h, 256, row0, k, lane);
    v16h a1 = frag_a(h, 256, row0 + 16, k, lane);
#pragma unroll
    for (int gq = 0; gq < 4; ++gq)
#pragma unroll
      for (int b = 0; b < 2; ++b) {
        v16h bf = frag_b_swz(Bs, 64, K0 + k, gq * 256 + j0 + b * 16, lane);
        acc[gq * 2 + b]     = WMMA(a0, bf, acc[gq * 2 + b]);
        acc[8 + gq * 2 + b] = WMMA(a1, bf, acc[8 + gq * 2 + b]);
      }
  }
  __syncthreads();  // all waves finished reading h before any wave overwrites it
  int g = (lane >> 4) & 1;
#pragma unroll
  for (int b = 0; b < 2; ++b) {
    int col = j0 + b * 16 + (lane & 15);
    float bI = b_ih[col] + b_hh[col];
    float bF = b_ih[256 + col] + b_hh[256 + col];
    float bG = b_ih[512 + col] + b_hh[512 + col];
    float bO = b_ih[768 + col] + b_hh[768 + col];
#pragma unroll
    for (int rt = 0; rt < 2; ++rt) {
#pragma unroll
      for (int v = 0; v < 8; ++v) {
        size_t idx = (size_t)(row0 + rt * 16 + v + 8 * g) * 256 + col;
        float I = 1.f / (1.f + __expf(-(acc[rt * 8 + 0 + b][v] + bI)));
        float F = 1.f / (1.f + __expf(-(acc[rt * 8 + 2 + b][v] + bF)));
        float G = tanhf(acc[rt * 8 + 4 + b][v] + bG);
        float O = 1.f / (1.f + __expf(-(acc[rt * 8 + 6 + b][v] + bO)));
        float cv = F * cst[idx] + I * G;
        cst[idx] = cv;
        h[idx] = (_Float16)(O * tanhf(cv));
      }
    }
  }
}

// ---------- flash MHA via S^T = K·Q^T: softmax in-lane, P^T repack in registers (no LDS) ----------
__global__ __launch_bounds__(256) void attention(const _Float16* __restrict__ qkv,
                                                 const _Float16* __restrict__ vt,
                                                 _Float16* __restrict__ ao) {
  int wave = threadIdx.x >> 5, lane = threadIdx.x & 31;
  int gw = blockIdx.x * 8 + wave;
  int head = gw >> 8;
  int qt = gw & 255;
  if (head >= NHEAD) return;
  int row0 = qt << 4;
  const _Float16* Qp = qkv + head * 32;
  const _Float16* Kp = qkv + 256 + head * 32;
  const _Float16* VtH = vt + (size_t)head * 32 * NN;
  v16h qb = frag_bt(Qp, 768, row0, lane);  // loop-invariant Q^T B-fragment (k=dh, n=query)
  float m = -1e30f, l = 0.f;
  v8f acc0 = {}, acc1 = {};
  const float scale = 0.17677669529663688f;  // 1/sqrt(32)
  for (int n0 = 0; n0 < NN; n0 += 32) {
    v16h ka0 = frag_a(Kp, 768, n0, 0, lane);
    v16h ka1 = frag_a(Kp, 768, n0 + 16, 0, lane);
    v8f st0 = {}, st1 = {};
    st0 = WMMA(ka0, qb, st0);  // S^T[key][query]: lane = query, VGPR = key
    st1 = WMMA(ka1, qb, st1);
    float cm = -1e30f;
#pragma unroll
    for (int v = 0; v < 8; ++v) {
      st0[v] *= scale;
      st1[v] *= scale;
      cm = fmaxf(cm, fmaxf(st0[v], st1[v]));
    }
    cm = fmaxf(cm, __shfl_xor(cm, 16, 32));
    float mn = fmaxf(m, cm);
    float al = __expf(m - mn);
    v16h pb;  // P^T B-fragment: f[j] = p0[j], f[8+j] = p1[j]  (exact D->B layout identity)
    float rs = 0.f;
#pragma unroll
    for (int v = 0; v < 8; ++v) {
      float p0 = __expf(st0[v] - mn);
      float p1 = __expf(st1[v] - mn);
      rs += p0 + p1;
      pb[v]     = (_Float16)p0;
      pb[8 + v] = (_Float16)p1;
    }
    rs += __shfl_xor(rs, 16, 32);
    l = l * al + rs;
#pragma unroll
    for (int v = 0; v < 8; ++v) { acc0[v] *= al; acc1[v] *= al; }
    m = mn;
    v16h va0 = frag_a(VtH, NN, 0, n0, lane);
    v16h va1 = frag_a(VtH, NN, 16, n0, lane);
    acc0 = WMMA(va0, pb, acc0);
    acc1 = WMMA(va1, pb, acc1);
  }
  float inv = 1.f / l;
  int q = lane & 15, g = (lane >> 4) & 1;
  v8h o0, o1;
#pragma unroll
  for (int v = 0; v < 8; ++v) {
    o0[v] = (_Float16)(acc0[v] * inv);
    o1[v] = (_Float16)(acc1[v] * inv);
  }
  _Float16* aop = ao + (size_t)(row0 + q) * 256 + head * 32;
  *(v8h*)(aop + 8 * g) = o0;
  *(v8h*)(aop + 16 + 8 * g) = o1;
}

// ---------- LayerNorm: one wave per row of 256 ----------
__global__ __launch_bounds__(256) void layernorm(const float* __restrict__ xin,
                                                 const float* __restrict__ w, const float* __restrict__ b,
                                                 _Float16* __restrict__ xo) {
  int wave = threadIdx.x >> 5, lane = threadIdx.x & 31;
  int row = blockIdx.x * 8 + wave;
  if (row >= NN) return;
  float vals[8];
  float s = 0.f;
#pragma unroll
  for (int j = 0; j < 8; ++j) { vals[j] = xin[(size_t)row * 256 + lane + 32 * j]; s += vals[j]; }
  for (int o = 1; o < 32; o <<= 1) s += __shfl_xor(s, o, 32);
  float mu = s * (1.f / 256.f);
  float vs = 0.f;
#pragma unroll
  for (int j = 0; j < 8; ++j) { float d = vals[j] - mu; vs += d * d; }
  for (int o = 1; o < 32; o <<= 1) vs += __shfl_xor(vs, o, 32);
  float inv = rsqrtf(vs * (1.f / 256.f) + 1e-5f);
#pragma unroll
  for (int j = 0; j < 8; ++j) {
    int ci = lane + 32 * j;
    xo[(size_t)row * 256 + ci] = (_Float16)((vals[j] - mu) * inv * w[ci] + b[ci]);
  }
}

// ---------- graph attention over edges ----------
__device__ __forceinline__ void atomicMaxF(float* a, float v) {
  if (v >= 0.f) atomicMax((int*)a, __float_as_int(v));
  else          atomicMin((unsigned int*)a, __float_as_uint(v));
}

__global__ __launch_bounds__(256) void edge_alpha(const float* __restrict__ qg, const float* __restrict__ kg,
                                                  const int* __restrict__ ei, float* __restrict__ alpha,
                                                  float* __restrict__ mbuf) {
  int idx = blockIdx.x * 256 + threadIdx.x;
  if (idx >= EE * HC_) return;
  int e = idx >> 3, h = idx & 7;
  int src = ei[e], dst = ei[EE + e];
  const float4* qp = (const float4*)(qg + (size_t)dst * 2048 + h * 256);
  const float4* kp = (const float4*)(kg + (size_t)src * 2048 + h * 256);
  float s = 0.f;
#pragma unroll 8
  for (int c4 = 0; c4 < 64; ++c4) {
    float4 a = qp[c4], b = kp[c4];
    s += a.x * b.x + a.y * b.y + a.z * b.z + a.w * b.w;
  }
  s *= (1.f / 16.f);
  alpha[idx] = s;
  atomicMaxF(&mbuf[dst * 8 + h], s);
}

__global__ __launch_bounds__(256) void edge_exp(const int* __restrict__ ei, float* __restrict__ alpha,
                                                const float* __restrict__ mbuf, float* __restrict__ sbuf) {
  int idx = blockIdx.x * 256 + threadIdx.x;
  if (idx >= EE * HC_) return;
  int e = idx >> 3, h = idx & 7;
  int dst = ei[EE + e];
  float ex = __expf(alpha[idx] - mbuf[dst * 8 + h]);
  alpha[idx] = ex;
  atomicAdd(&sbuf[dst * 8 + h], ex);
}

__global__ __launch_bounds__(256) void edge_msg(const int* __restrict__ ei, const float* __restrict__ alpha,
                                                const float* __restrict__ sbuf, const float* __restrict__ vg,
                                                float* __restrict__ agg) {
  int e = blockIdx.x;
  int c = threadIdx.x;
  int src = ei[e], dst = ei[EE + e];
#pragma unroll
  for (int h = 0; h < 8; ++h) {
    float a = alpha[e * 8 + h] / (sbuf[dst * 8 + h] + 1e-16f);
    atomicAdd(&agg[(size_t)dst * 2048 + h * 256 + c], vg[(size_t)src * 2048 + h * 256 + c] * a);
  }
}

__global__ __launch_bounds__(256) void final_out(const float* __restrict__ agg, const float* __restrict__ skipo,
                                                 float* __restrict__ out) {
  size_t idx = (size_t)blockIdx.x * 256 + threadIdx.x;
  if (idx >= (size_t)NN * 256) return;
  size_t n = idx >> 8;
  int c = (int)(idx & 255);
  float s = 0.f;
#pragma unroll
  for (int h = 0; h < 8; ++h) s += agg[n * 2048 + h * 256 + c];
  out[idx] = s * 0.125f + skipo[idx];
}

// ==================== host-side orchestration ====================
extern "C" void kernel_launch(void* const* d_in, const int* in_sizes, int n_in,
                              void* d_out, int out_size, void* d_ws, size_t ws_size,
                              hipStream_t stream) {
  const float* x      = (const float*)d_in[0];
  const int*   ei     = (const int*)d_in[1];
  const float* W_ih0  = (const float*)d_in[2];
  const float* W_hh0  = (const float*)d_in[3];
  const float* b_ih0  = (const float*)d_in[4];
  const float* b_hh0  = (const float*)d_in[5];
  const float* W_ih1  = (const float*)d_in[6];
  const float* W_hh1  = (const float*)d_in[7];
  const float* b_ih1  = (const float*)d_in[8];
  const float* b_hh1  = (const float*)d_in[9];
  const float* in_pw  = (const float*)d_in[10];
  const float* in_pb  = (const float*)d_in[11];
  const float* out_pw = (const float*)d_in[12];
  const float* out_pb = (const float*)d_in[13];
  const float* ln_w   = (const float*)d_in[14];
  const float* ln_b   = (const float*)d_in[15];
  const float* Wq     = (const float*)d_in[16];
  const float* bq     = (const float*)d_in[17];
  const float* Wk     = (const float*)d_in[18];
  const float* bk     = (const float*)d_in[19];
  const float* Wv     = (const float*)d_in[20];
  const float* bv     = (const float*)d_in[21];
  const float* Wskip  = (const float*)d_in[22];
  const float* bskip  = (const float*)d_in[23];
  float* out = (float*)d_out;

  char* p = (char*)d_ws;
  auto take = [&](size_t bytes) { char* r = p; p += (bytes + 255) & ~(size_t)255; return r; };

  _Float16* xpad  = (_Float16*)take((size_t)TT * NN * 32 * 2);
  _Float16* B0    = (_Float16*)take((size_t)288 * 1024 * 2);
  _Float16* B1    = (_Float16*)take((size_t)512 * 1024 * 2);
  _Float16* Bqkv  = (_Float16*)take((size_t)256 * 768 * 2);
  _Float16* Bout  = (_Float16*)take((size_t)256 * 256 * 2);
  _Float16* Bq    = (_Float16*)take((size_t)256 * 2048 * 2);
  _Float16* Bk    = (_Float16*)take((size_t)256 * 2048 * 2);
  _Float16* Bv    = (_Float16*)take((size_t)256 * 2048 * 2);
  _Float16* Bskip = (_Float16*)take((size_t)256 * 256 * 2);
  _Float16* h0    = (_Float16*)take((size_t)NN * 256 * 2);
  float*    c0    = (float*)take((size_t)NN * 256 * 4);
  _Float16* h1    = (_Float16*)take((size_t)NN * 256 * 2);
  float*    c1    = (float*)take((size_t)NN * 256 * 4);
  _Float16* qkv16 = (_Float16*)take((size_t)NN * 768 * 2);
  _Float16* vt    = (_Float16*)take((size_t)256 * NN * 2);
  _Float16* ao16  = (_Float16*)take((size_t)NN * 256 * 2);
  float*    attnb = (float*)take((size_t)NN * 256 * 4);
  _Float16* xn16  = (_Float16*)take((size_t)NN * 256 * 2);
  float*    qg    = (float*)take((size_t)NN * 2048 * 4);
  float*    kg    = (float*)take((size_t)NN * 2048 * 4);
  float*    vg    = (float*)take((size_t)NN * 2048 * 4);
  float*    alpha = (float*)take((size_t)EE * 8 * 4);
  float*    mbuf  = (float*)take((size_t)NN * 8 * 4);
  float*    sbuf  = (float*)take((size_t)NN * 8 * 4);
  float*    agg   = (float*)take((size_t)NN * 2048 * 4);
  float*    skipo = (float*)take((size_t)NN * 256 * 4);
  (void)ws_size; (void)in_sizes; (void)n_in; (void)out_size;

  // --- weight prep ---
  fill_f16<<<512, 256, 0, stream>>>(B0, (size_t)288 * 1024);
  wtrans_swz<<<256, 256, 0, stream>>>(W_ih0, B0, 1024, 16, 64, 0);
  wtrans_swz<<<512, 256, 0, stream>>>(W_hh0, B0, 1024, 256, 64, 32);
  wtrans_swz<<<512, 256, 0, stream>>>(W_ih1, B1, 1024, 256, 64, 0);
  wtrans_swz<<<512, 256, 0, stream>>>(W_hh1, B1, 1024, 256, 64, 256);
  wtrans_swz<<<512, 256, 0, stream>>>(in_pw, Bqkv, 768, 256, 48, 0);
  wtrans_swz<<<256, 256, 0, stream>>>(out_pw, Bout, 256, 256, 16, 0);
  wtrans_swz<<<512, 256, 0, stream>>>(Wq, Bq, 2048, 256, 128, 0);
  wtrans_swz<<<512, 256, 0, stream>>>(Wk, Bk, 2048, 256, 128, 0);
  wtrans_swz<<<512, 256, 0, stream>>>(Wv, Bv, 2048, 256, 128, 0);
  wtrans_swz<<<256, 256, 0, stream>>>(Wskip, Bskip, 256, 256, 16, 0);
  xpad_k<<<1024, 256, 0, stream>>>(x, xpad);

  // --- LSTM state init ---
  fill_f16<<<1024, 256, 0, stream>>>(h0, (size_t)NN * 256);
  fill_f32<<<1024, 256, 0, stream>>>(c0, 0.f, (size_t)NN * 256);
  fill_f16<<<1024, 256, 0, stream>>>(h1, (size_t)NN * 256);
  fill_f32<<<1024, 256, 0, stream>>>(c1, 0.f, (size_t)NN * 256);

  // --- 2-layer LSTM over T steps (WG = 32 rows, double row-tile B reuse) ---
  for (int t = 0; t < TT; ++t) {
    lstm_cell<32><<<NN / 32, 256, 0, stream>>>(xpad + (size_t)t * NN * 32, 32, B0, b_ih0, b_hh0, c0, h0);
    lstm_cell<256><<<NN / 32, 256, 0, stream>>>(h0, 256, B1, b_ih1, b_hh1, c1, h1);
  }

  // --- qkv projection -> f16, V transpose ---
  gemm_wmma<256, false, true><<<(256 * 48 + 7) / 8, 256, 0, stream>>>(h1, Bqkv, in_pb, nullptr, qkv16, NN, 768);
  vtrans<<<1024, 256, 0, stream>>>(qkv16, vt);

  // --- multi-head attention ---
  attention<<<256, 256, 0, stream>>>(qkv16, vt, ao16);

  // --- out_proj -> f32 ---
  gemm_wmma<256, true, false><<<(256 * 16 + 7) / 8, 256, 0, stream>>>(ao16, Bout, out_pb, attnb, nullptr, NN, 256);

  // --- layernorm -> xn (f16) ---
  layernorm<<<512, 256, 0, stream>>>(attnb, ln_w, ln_b, xn16);

  // --- q/k/v graph projections -> f32 ---
  gemm_wmma<256, true, false><<<(256 * 128 + 7) / 8, 256, 0, stream>>>(xn16, Bq, bq, qg, nullptr, NN, 2048);
  gemm_wmma<256, true, false><<<(256 * 128 + 7) / 8, 256, 0, stream>>>(xn16, Bk, bk, kg, nullptr, NN, 2048);
  gemm_wmma<256, true, false><<<(256 * 128 + 7) / 8, 256, 0, stream>>>(xn16, Bv, bv, vg, nullptr, NN, 2048);

  // --- edge softmax + aggregation ---
  fill_f32<<<256, 256, 0, stream>>>(mbuf, -__builtin_inff(), (size_t)NN * 8);
  fill_f32<<<256, 256, 0, stream>>>(sbuf, 0.f, (size_t)NN * 8);
  fill_f32<<<2048, 256, 0, stream>>>(agg, 0.f, (size_t)NN * 2048);
  edge_alpha<<<(EE * HC_) / 256, 256, 0, stream>>>(qg, kg, ei, alpha, mbuf);
  edge_exp<<<(EE * HC_) / 256, 256, 0, stream>>>(ei, alpha, mbuf, sbuf);
  edge_msg<<<EE, 256, 0, stream>>>(ei, alpha, sbuf, vg, agg);

  // --- skip GEMM + final combine ---
  gemm_wmma<256, true, false><<<(256 * 16 + 7) / 8, 256, 0, stream>>>(xn16, Bskip, bskip, skipo, nullptr, NN, 256);
  final_out<<<NN, 256, 0, stream>>>(agg, skipo, out);
}
